// SimpleGCN_42314017800420
// MI455X (gfx1250) — matrix operand order repaired and checked by reading, not compiled, not run on hardware
//
#include <hip/hip_runtime.h>
#include <hip/hip_bf16.h>

#define DD 256   // feature dimension

typedef __attribute__((ext_vector_type(16))) __bf16 v16bf;
typedef __attribute__((ext_vector_type(8)))  __bf16 v8bf;
typedef __attribute__((ext_vector_type(8)))  float  v8f;

// ---------------------------------------------------------------- utilities

__global__ void gcn_fill_f32(float* __restrict__ p, float v, int n) {
    int i = blockIdx.x * blockDim.x + threadIdx.x;
    if (i < n) p[i] = v;
}

__global__ void gcn_degree(const int* __restrict__ dst, float* __restrict__ deg, int n_edges) {
    int e = blockIdx.x * blockDim.x + threadIdx.x;
    if (e < n_edges) atomicAdd(&deg[dst[e]], 1.0f);
}

__global__ void gcn_rsqrt(float* __restrict__ d, int n) {
    int i = blockIdx.x * blockDim.x + threadIdx.x;
    if (i < n) d[i] = rsqrtf(d[i]);
}

__device__ __forceinline__ unsigned gcn_hash(unsigned v) {
    v ^= 0x9e3779b9u; v *= 0x85ebca6bu; v ^= v >> 13; v *= 0xc2b2ae35u; v ^= v >> 16;
    return v;
}

// f32 -> bf16 bulk convert (X is converted once; cost ~77MB of traffic, noise)
__global__ void gcn_f32_to_bf16(const float* __restrict__ s, __bf16* __restrict__ d, int n) {
    int i = blockIdx.x * blockDim.x + threadIdx.x;
    if (i < n) d[i] = (__bf16)s[i];
}

// Pack W (256x256 f32, row-major [k][n]) into fragment-ready bf16:
//   Wp[((k>>5)*256 + n)*32 + (k&31 with bits 3,4 swapped)]  -> each lane's B fragment is one
//   contiguous 32-byte read (2 x global_load_b128).
__global__ void gcn_pack_w(const float* __restrict__ W, __bf16* __restrict__ Wp) {
    int i = blockIdx.x * blockDim.x + threadIdx.x;   // 65536
    if (i >= DD * DD) return;
    int k = i >> 8, n = i & 255;
    Wp[(size_t)(((k >> 5) * DD + n) * 32 + (((k >> 3) & 1) * 16 + ((k >> 4) & 1) * 8 + (k & 7)))] = (__bf16)W[i];
}

// ------------------------------------------------- GEMM: H = Xb @ W + b (WMMA)
// Block = 512 threads = 16 waves; one 16-row strip of H per block, one 16x16
// N-tile per wave. The 8KB bf16 A tile is staged via CDNA5 async global->LDS
// (global_load_async_to_lds_b128, drained with s_wait_asynccnt), then read as
// ds_load_b128 fragments. B comes from the packed bf16 W image (L2-resident).

__global__ __launch_bounds__(512)
void gcn_gemm_bias_wmma(const __bf16* __restrict__ Xb, const v16bf* __restrict__ Wp,
                        const float* __restrict__ bias, float* __restrict__ H, int n_rows) {
    __shared__ __bf16 sA[16 * DD];                 // 8 KB tile
    const int row0 = blockIdx.x * 16;

    {   // async copy: 512 threads x 16B chunks = 8KB (1 async op per wave)
        int id = threadIdx.x;
        int r  = id >> 5;                           // 32 chunks per 512B row
        int gr = row0 + r; if (gr >= n_rows) gr = n_rows - 1;   // clamp (safe: extra rows never stored)
        int cb = (id & 31) * 16;                    // byte offset within row
        unsigned lds = (unsigned)(size_t)(&sA[0]) + (unsigned)(r * (DD * 2) + cb);
        unsigned gof = (unsigned)((size_t)gr * (DD * 2) + (size_t)cb);
        asm volatile("global_load_async_to_lds_b128 %0, %1, %2 offset:0"
                     :: "v"(lds), "v"(gof), "s"(Xb) : "memory");
    }
    asm volatile("s_wait_asynccnt 0" ::: "memory");
    __syncthreads();

    const int lane  = threadIdx.x & 31;            // wave32
    const int wave  = threadIdx.x >> 5;
    // A fragment (16-bit A 16x32, ISA 7.12.2): per lane two contiguous
    // 8-element runs at K = akoff and K = akoff+16 within the 32-K block.
    const int arow  = lane & 15;
    const int akoff = (lane >> 4) * 8;
    // B fragment: column per lane, K halves split across lane halves.
    const int bcol  = wave * 16 + (lane & 15);
    const int bsel  = lane >> 4;

    v8f acc = {};
#pragma unroll
    for (int kb = 0; kb < DD; kb += 32) {
        const __bf16* ap = &sA[arow * DD + kb + akoff];
        union { v16bf v; v8bf h[2]; } A;
        A.h[0] = *(const v8bf*)(ap);               // ds_load_b128
        A.h[1] = *(const v8bf*)(ap + 16);          // ds_load_b128
        v16bf B = Wp[(size_t)(((kb >> 5) * DD + bcol) * 2 + bsel)];  // 2x global_load_b128
        acc = __builtin_amdgcn_wmma_f32_16x16x32_bf16(
            /*neg_a=*/false, A.v, /*neg_b=*/false, B,
            /*c_mod=*/(short)0, acc, /*reuse_a=*/false, /*reuse_b=*/false);
    }

    union { v8f v; float e[8]; } C; C.v = acc;
    const float bb = bias[bcol];
    // C/D layout: M = vgpr + 8*(lane>=16), N = lane&15
#pragma unroll
    for (int vi = 0; vi < 8; ++vi) {
        int m = row0 + vi + ((lane >> 4) * 8);
        if (m < n_rows) H[(size_t)m * DD + bcol] = C.e[vi] + bb;
    }
}

// ------------------------------------------------- edge gather + scatter-add
// 64 lanes (2 waves) per edge, float4 per lane: AGG[dst] += H[src] * norm.
// AGG (51 MB) is L2-resident, so the f32 atomics resolve in L2.

__global__ __launch_bounds__(256)
void gcn_edge_scatter(const int* __restrict__ src, const int* __restrict__ dst,
                      const float* __restrict__ dis, const float* __restrict__ H,
                      float* __restrict__ AGG, int n_edges) {
    int e = blockIdx.x * 4 + (threadIdx.x >> 6);
    if (e >= n_edges) return;
    int s = src[e], d = dst[e];
    float norm = dis[s] * dis[d];
    int c = (threadIdx.x & 63) * 4;
    const float4 v = *reinterpret_cast<const float4*>(H + (size_t)s * DD + c);
    float* a = AGG + (size_t)d * DD + c;
    atomicAdd(a + 0, v.x * norm);
    atomicAdd(a + 1, v.y * norm);
    atomicAdd(a + 2, v.z * norm);
    atomicAdd(a + 3, v.w * norm);
}

// layer-1 epilogue: self-loop + relu + dropout(p=0.5, scale 2).
// Output goes straight to bf16 (its only consumer is the layer-2 WMMA GEMM).
__global__ void gcn_combine1(const float* __restrict__ AGG, const float* __restrict__ H,
                             const float* __restrict__ dis, __bf16* __restrict__ Hb, int total) {
    int i = blockIdx.x * blockDim.x + threadIdx.x;
    if (i >= total) return;
    int node = i >> 8;
    float sn  = dis[node] * dis[node];
    float val = fmaxf(AGG[i] + H[i] * sn, 0.0f);
    unsigned keep = gcn_hash((unsigned)i ^ 42u) & 1u;
    Hb[i] = (__bf16)(keep ? val * 2.0f : 0.0f);
}

// layer-2 epilogue fused with mean-pool numerator:
// out[batch[node]] += relu(agg2 + h2*self_norm)
__global__ void gcn_combine2_pool(const float* __restrict__ AGG2, const float* __restrict__ H2,
                                  const float* __restrict__ dis, const int* __restrict__ batch,
                                  float* __restrict__ out, int total) {
    int i = blockIdx.x * blockDim.x + threadIdx.x;
    if (i >= total) return;
    int node = i >> 8;
    int c    = i & 255;
    float sn  = dis[node] * dis[node];
    float val = fmaxf(AGG2[i] + H2[i] * sn, 0.0f);
    atomicAdd(&out[(size_t)batch[node] * DD + c], val);
}

__global__ void gcn_count(const int* __restrict__ batch, float* __restrict__ counts, int n) {
    int i = blockIdx.x * blockDim.x + threadIdx.x;
    if (i < n) atomicAdd(&counts[batch[i]], 1.0f);
}

__global__ void gcn_divide(float* __restrict__ out, const float* __restrict__ counts, int total) {
    int i = blockIdx.x * blockDim.x + threadIdx.x;
    if (i < total) out[i] /= fmaxf(counts[i >> 8], 1.0f);
}

// ---------------------------------------------------------------- launcher

extern "C" void kernel_launch(void* const* d_in, const int* in_sizes, int n_in,
                              void* d_out, int out_size, void* d_ws, size_t ws_size,
                              hipStream_t stream) {
    const float* x     = (const float*)d_in[0];
    const int*   ei    = (const int*)d_in[1];
    const int*   batch = (const int*)d_in[2];
    const float* W1    = (const float*)d_in[3];
    const float* b1    = (const float*)d_in[4];
    const float* W2    = (const float*)d_in[5];
    const float* b2    = (const float*)d_in[6];

    const int n_nodes    = in_sizes[0] / DD;
    const int n_edges    = in_sizes[1] / 2;
    const int num_graphs = out_size / DD;
    const int* src  = ei;
    const int* dstp = ei + n_edges;
    float* out = (float*)d_out;

    // workspace carve-up (float units, each region 256B-aligned):
    // dis | counts | H (f32) | AGG (f32) | XHb (bf16, shared X->bf16 / layer1 acts) | Wp1 | Wp2
    float* ws = (float*)d_ws;
    size_t o = 0;
    float* dis    = ws + o; o += (size_t)n_nodes;
    float* counts = ws + o; o += (size_t)num_graphs;
    o = (o + 63) & ~(size_t)63;
    float*  H   = ws + o; o += (size_t)n_nodes * DD;
    float*  AGG = ws + o; o += (size_t)n_nodes * DD;
    __bf16* XHb = (__bf16*)(ws + o); o += (size_t)n_nodes * DD / 2;   // bf16, half the floats
    o = (o + 63) & ~(size_t)63;
    __bf16* Wp1 = (__bf16*)(ws + o); o += (size_t)DD * DD / 2;
    o = (o + 63) & ~(size_t)63;
    __bf16* Wp2 = (__bf16*)(ws + o); o += (size_t)DD * DD / 2;
    (void)ws_size; (void)n_in;

    const int total = n_nodes * DD;   // 12.8M, fits int32
    auto cdiv = [](int a, int b) { return (unsigned)((a + b - 1) / b); };
    dim3 b256(256);

    // init: deg=1 (self-loop), counts=0, AGG=0, out=0
    gcn_fill_f32<<<cdiv(n_nodes, 256), b256, 0, stream>>>(dis, 1.0f, n_nodes);
    gcn_fill_f32<<<cdiv(num_graphs, 256), b256, 0, stream>>>(counts, 0.0f, num_graphs);
    gcn_fill_f32<<<cdiv(total, 256), b256, 0, stream>>>(AGG, 0.0f, total);
    gcn_fill_f32<<<cdiv(out_size, 256), b256, 0, stream>>>(out, 0.0f, out_size);

    // weights -> fragment-ready bf16 images; X -> bf16
    gcn_pack_w<<<cdiv(DD * DD, 256), b256, 0, stream>>>(W1, Wp1);
    gcn_pack_w<<<cdiv(DD * DD, 256), b256, 0, stream>>>(W2, Wp2);
    gcn_f32_to_bf16<<<cdiv(total, 256), b256, 0, stream>>>(x, XHb, total);

    // degrees -> D^{-1/2}
    gcn_degree<<<cdiv(n_edges, 256), b256, 0, stream>>>(dstp, dis, n_edges);
    gcn_rsqrt<<<cdiv(n_nodes, 256), b256, 0, stream>>>(dis, n_nodes);

    // layer 1
    gcn_gemm_bias_wmma<<<cdiv(n_nodes, 16), dim3(512), 0, stream>>>(XHb, (const v16bf*)Wp1, b1, H, n_nodes);
    gcn_edge_scatter<<<cdiv(n_edges, 4), b256, 0, stream>>>(src, dstp, dis, H, AGG, n_edges);
    gcn_combine1<<<cdiv(total, 256), b256, 0, stream>>>(AGG, H, dis, XHb, total);  // XHb now holds layer-1 acts (bf16)

    // layer 2 (AGG <- h2; H reused as agg2)
    gcn_gemm_bias_wmma<<<cdiv(n_nodes, 16), dim3(512), 0, stream>>>(XHb, (const v16bf*)Wp2, b2, AGG, n_nodes);
    gcn_fill_f32<<<cdiv(total, 256), b256, 0, stream>>>(H, 0.0f, total);
    gcn_edge_scatter<<<cdiv(n_edges, 4), b256, 0, stream>>>(src, dstp, dis, AGG, H, n_edges);

    // mean pool
    gcn_count<<<cdiv(n_nodes, 256), b256, 0, stream>>>(batch, counts, n_nodes);
    gcn_combine2_pool<<<cdiv(total, 256), b256, 0, stream>>>(H, AGG, dis, batch, out, total);
    gcn_divide<<<cdiv(out_size, 256), b256, 0, stream>>>(out, counts, out_size);
}
